// MeshGraphNet_56487409877737
// MI455X (gfx1250) — matrix operand order repaired and checked by reading, not compiled, not run on hardware
//
#include <hip/hip_runtime.h>
#include <math.h>

// ---------------------------------------------------------------------------
// MeshGraphNet forward for MI455X (gfx1250, wave32, WMMA bf16->f32).
//
// Precision: activations f32 in HBM (residual/segment-sum fidelity),
// matmuls in bf16 via v_wmma_f32_16x16x32_bf16 (f32 accumulators).
// Block MLPs process 32 rows per wave so each B-fragment (weight tile) load
// feeds two WMMAs -> halves L2 weight traffic (which otherwise exceeds the
// HBM activation traffic by ~4x).
// ---------------------------------------------------------------------------

typedef __attribute__((ext_vector_type(16))) __bf16 v16bf;
typedef __attribute__((ext_vector_type(8)))  float  v8f;

#define EPSF 1e-6f

__device__ __forceinline__ v8f wmma_bf16(v16bf a, v16bf b, v8f c) {
  // D = A(16x32 bf16) x B(32x16 bf16) + C(16x16 f32)
  return __builtin_amdgcn_wmma_f32_16x16x32_bf16(false, a, false, b, (short)0, c,
                                                 false, false);
}

// Build an A-fragment (ISA 7.12.2, 16-bit A 16x32):
//   lanes 0-15 : row M=lane,    elems j<8 -> K=j,   j>=8 -> K=16+(j-8)
//   lanes 16-31: row M=lane-16, elems j<8 -> K=8+j, j>=8 -> K=24+(j-8)
// p = row base + 32*chunk; hoff = (lane>=16)?8:0
__device__ __forceinline__ v16bf load_afrag(const float* __restrict__ p, int hoff) {
  v16bf a;
#pragma unroll
  for (int j = 0; j < 8; ++j) a[j] = (__bf16)p[hoff + j];
#pragma unroll
  for (int j = 0; j < 8; ++j) a[8 + j] = (__bf16)p[16 + hoff + j];
  return a;
}

// ---------------------------------------------------------------------------
// Weight packing: W (Kact x 128, row-major f32) -> bf16 B-fragments.
// B layout (dense 16-bit 32x16): lane<16: col=lane, K=c*32+{0..7,16..23};
// lane>=16: col=lane-16, K=c*32+{8..15,24..31}. Each lane slot = 16 bf16 contiguous.
// ---------------------------------------------------------------------------
__global__ void mgn_pack_w(const float* __restrict__ W, __bf16* __restrict__ out,
                           int Kpad, int Kact) {
  const int t = blockIdx.x * blockDim.x + threadIdx.x;
  const int nslots = (Kpad / 32) * 8 * 32;
  if (t >= nslots) return;
  const int f = t >> 5;       // fragment index = c*8+n
  const int lane = t & 31;
  const int c = f >> 3, n = f & 7;
  const int colN = n * 16 + (lane & 15);
  const int kbase = c * 32 + (lane >> 4) * 8;
  __bf16* o = out + (size_t)t * 16;
#pragma unroll
  for (int j = 0; j < 16; ++j) {
    const int k = kbase + (j < 8 ? j : j + 8);
    o[j] = (__bf16)((k < Kact) ? W[(size_t)k * 128 + colN] : 0.f);
  }
}

// ---------------------------------------------------------------------------
// Feature-prep kernels (write 32-wide zero-padded f32 rows)
// ---------------------------------------------------------------------------
__global__ void mgn_feat_node(const float* __restrict__ vel,
                              const float* __restrict__ imass,
                              const float* __restrict__ nattr,
                              const float* __restrict__ nmean, const float* __restrict__ nstd,
                              const float* __restrict__ mmean, const float* __restrict__ mstd,
                              float* __restrict__ o32, int N) {
  const int i = blockIdx.x * blockDim.x + threadIdx.x;
  if (i >= N) return;
  float* o = o32 + (size_t)i * 32;
#pragma unroll
  for (int k = 0; k < 4; ++k)
    o[k] = (vel[(size_t)i * 4 + k] - nmean[k]) / (nstd[k] + EPSF);
  const float im = imass[i];
  const float sg = (im > 0.f) ? 1.f : ((im < 0.f) ? -1.f : 0.f);
  const float m = sg * log1pf(fabsf(im));
  o[4] = (m - mmean[0]) / (mstd[0] + EPSF);
  o[5] = nattr[(size_t)i * 2 + 0];
  o[6] = nattr[(size_t)i * 2 + 1];
#pragma unroll
  for (int k = 7; k < 32; ++k) o[k] = 0.f;
}

__global__ void mgn_feat_mesh(const float* __restrict__ pos,
                              const float* __restrict__ mpos,
                              const float* __restrict__ vel,
                              const int* __restrict__ src, const int* __restrict__ dst,
                              const float* __restrict__ mean, const float* __restrict__ stdv,
                              float* __restrict__ o32, int E) {
  const int e = blockIdx.x * blockDim.x + threadIdx.x;
  if (e >= E) return;
  const int s = src[e], d = dst[e];
  const float rx = pos[(size_t)s * 2] - pos[(size_t)d * 2];
  const float ry = pos[(size_t)s * 2 + 1] - pos[(size_t)d * 2 + 1];
  const float dist = sqrtf(rx * rx + ry * ry);
  const float sx = mpos[(size_t)s * 2] - mpos[(size_t)d * 2];
  const float sy = mpos[(size_t)s * 2 + 1] - mpos[(size_t)d * 2 + 1];
  const float dist_s = sqrtf(sx * sx + sy * sy);
  const float strain = (dist - dist_s) / (dist_s + EPSF);
  const float svx = strain * (rx / dist + EPSF);
  const float svy = strain * (ry / dist + EPSF);
  const float rvx = vel[(size_t)s * 4 + 2] - vel[(size_t)d * 4 + 2];
  const float rvy = vel[(size_t)s * 4 + 3] - vel[(size_t)d * 4 + 3];
  const float raw[8] = {rx, ry, svx, svy, rvx, rvy, dist, dist_s};
  float* o = o32 + (size_t)e * 32;
#pragma unroll
  for (int k = 0; k < 8; ++k) o[k] = (raw[k] - mean[k]) / (stdv[k] + EPSF);
#pragma unroll
  for (int k = 8; k < 32; ++k) o[k] = 0.f;
}

__global__ void mgn_feat_world(const float* __restrict__ pos,
                               const float* __restrict__ vel,
                               const int* __restrict__ sw, const int* __restrict__ dw,
                               const float* __restrict__ mean, const float* __restrict__ stdv,
                               float* __restrict__ o32, int EW) {
  const int e = blockIdx.x * blockDim.x + threadIdx.x;
  if (e >= EW) return;
  const int s = sw[e], d = dw[e];
  const float raw[4] = {pos[(size_t)s * 2] - pos[(size_t)d * 2],
                        pos[(size_t)s * 2 + 1] - pos[(size_t)d * 2 + 1],
                        vel[(size_t)s * 4 + 2] - vel[(size_t)d * 4 + 2],
                        vel[(size_t)s * 4 + 3] - vel[(size_t)d * 4 + 3]};
  float* o = o32 + (size_t)e * 32;
#pragma unroll
  for (int k = 0; k < 4; ++k) o[k] = (raw[k] - mean[k]) / (stdv[k] + EPSF);
#pragma unroll
  for (int k = 4; k < 32; ++k) o[k] = 0.f;
}

__global__ void mgn_zero2(float* __restrict__ a, float* __restrict__ b, int n) {
  const int i = blockIdx.x * blockDim.x + threadIdx.x;
  if (i < n) { a[i] = 0.f; b[i] = 0.f; }
}

// ---------------------------------------------------------------------------
// Encoder MLP: [pad32 -> 128 -> 128 -> 128] + LayerNorm. One wave = 16 rows.
// ---------------------------------------------------------------------------
__global__ void __launch_bounds__(128)
mgn_enc_mlp(const float* __restrict__ in32,
            const v16bf* __restrict__ W0, const v16bf* __restrict__ W1,
            const v16bf* __restrict__ W2,
            const float* __restrict__ b0, const float* __restrict__ b1,
            const float* __restrict__ b2,
            const float* __restrict__ lnsc, const float* __restrict__ lnbi,
            float* __restrict__ out, int rows) {
  __shared__ float lds[4][16 * 132];
  const int wave = threadIdx.x >> 5, lane = threadIdx.x & 31;
  const int tile = blockIdx.x * 4 + wave;
  const int r0 = tile * 16;
  if (r0 >= rows) return;
  const int half = lane >> 4, lcol = lane & 15, hoff = half * 8;
  int arow = r0 + lcol; if (arow >= rows) arow = rows - 1;
  float* hl = &lds[wave][0];

  // ---- layer 0: K=32 (padded) ----
  v8f acc[8];
#pragma unroll
  for (int n = 0; n < 8; ++n) {
    const float bv = b0[n * 16 + lcol];
#pragma unroll
    for (int v = 0; v < 8; ++v) acc[n][v] = bv;
  }
  {
    const v16bf a = load_afrag(in32 + (size_t)arow * 32, hoff);
#pragma unroll
    for (int n = 0; n < 8; ++n) acc[n] = wmma_bf16(a, W0[n * 32 + lane], acc[n]);
  }
#pragma unroll
  for (int n = 0; n < 8; ++n)
#pragma unroll
    for (int v = 0; v < 8; ++v)
      hl[(v + half * 8) * 132 + n * 16 + lcol] = fmaxf(acc[n][v], 0.f);

  // ---- layer 1: 128 -> 128 ----
#pragma unroll
  for (int n = 0; n < 8; ++n) {
    const float bv = b1[n * 16 + lcol];
#pragma unroll
    for (int v = 0; v < 8; ++v) acc[n][v] = bv;
  }
#pragma unroll
  for (int c = 0; c < 4; ++c) {
    const v16bf a = load_afrag(hl + lcol * 132 + c * 32, hoff);
#pragma unroll
    for (int n = 0; n < 8; ++n)
      acc[n] = wmma_bf16(a, W1[(c * 8 + n) * 32 + lane], acc[n]);
  }
#pragma unroll
  for (int n = 0; n < 8; ++n)
#pragma unroll
    for (int v = 0; v < 8; ++v)
      hl[(v + half * 8) * 132 + n * 16 + lcol] = fmaxf(acc[n][v], 0.f);

  // ---- layer 2: 128 -> 128 (no relu) ----
  v8f acc2[8];
#pragma unroll
  for (int n = 0; n < 8; ++n) {
    const float bv = b2[n * 16 + lcol];
#pragma unroll
    for (int v = 0; v < 8; ++v) acc2[n][v] = bv;
  }
#pragma unroll
  for (int c = 0; c < 4; ++c) {
    const v16bf a = load_afrag(hl + lcol * 132 + c * 32, hoff);
#pragma unroll
    for (int n = 0; n < 8; ++n)
      acc2[n] = wmma_bf16(a, W2[(c * 8 + n) * 32 + lane], acc2[n]);
  }

  // ---- layernorm + store ----
  float mu[8], rs[8];
#pragma unroll
  for (int v = 0; v < 8; ++v) {
    float s = 0.f, s2 = 0.f;
#pragma unroll
    for (int n = 0; n < 8; ++n) { const float y = acc2[n][v]; s += y; s2 += y * y; }
#pragma unroll
    for (int m = 1; m <= 8; m <<= 1) { s += __shfl_xor(s, m, 32); s2 += __shfl_xor(s2, m, 32); }
    const float m1 = s * (1.f / 128.f);
    mu[v] = m1;
    rs[v] = rsqrtf(fmaxf(s2 * (1.f / 128.f) - m1 * m1, 0.f) + 1e-5f);
  }
  float lsc[8], lbi[8];
#pragma unroll
  for (int n = 0; n < 8; ++n) { lsc[n] = lnsc[n * 16 + lcol]; lbi[n] = lnbi[n * 16 + lcol]; }
#pragma unroll
  for (int v = 0; v < 8; ++v) {
    const int grow = r0 + v + half * 8;
    if (grow >= rows) continue;
#pragma unroll
    for (int n = 0; n < 8; ++n)
      out[(size_t)grow * 128 + n * 16 + lcol] =
          (acc2[n][v] - mu[v]) * rs[v] * lsc[n] + lbi[n];
  }
}

// ---------------------------------------------------------------------------
// Block MLP: concat(in0[g0], in1[g1], in2[row]) (384) -> 128 -> 128, LN,
// residual add, optional fused atomic segment-sum into agg[aggidx[row]].
// One wave = 32 rows (two 16-row subtiles sharing each B-fragment load).
// Workgroup = 2 waves (64 threads) so 32x132 f32/wave staging fits in LDS.
// ---------------------------------------------------------------------------
__global__ void __launch_bounds__(64)
mgn_block_mlp(const float* __restrict__ in0, const int* __restrict__ idx0,
              const float* __restrict__ in1, const int* __restrict__ idx1,
              const float* __restrict__ in2,
              const float* __restrict__ resid,
              float* __restrict__ out,
              float* __restrict__ agg, const int* __restrict__ aggidx,
              const v16bf* __restrict__ W1, const v16bf* __restrict__ W2,
              const float* __restrict__ bias1, const float* __restrict__ bias2,
              const float* __restrict__ lnsc, const float* __restrict__ lnbi,
              int rows) {
  __shared__ float lds[2][32 * 132];
  const int wave = threadIdx.x >> 5, lane = threadIdx.x & 31;
  const int tile = blockIdx.x * 2 + wave;
  const int r0 = tile * 32;
  if (r0 >= rows) return;
  const int half = lane >> 4, lcol = lane & 15, hoff = half * 8;
  float* hl = &lds[wave][0];

  const float* rp[2][3];
#pragma unroll
  for (int s = 0; s < 2; ++s) {
    int arow = r0 + s * 16 + lcol;
    if (arow >= rows) arow = rows - 1;
    const int g0 = idx0 ? idx0[arow] : arow;
    const int g1 = idx1 ? idx1[arow] : arow;
    rp[s][0] = in0 + (size_t)g0 * 128;
    rp[s][1] = in1 + (size_t)g1 * 128;
    rp[s][2] = in2 + (size_t)arow * 128;
  }

  // ---- layer 1: 384 -> 128; 12 K-chunks x 8 N-tiles, B reused for 2 tiles --
  v8f acc[2][8];
#pragma unroll
  for (int n = 0; n < 8; ++n) {
    const float bv = bias1[n * 16 + lcol];
#pragma unroll
    for (int v = 0; v < 8; ++v) { acc[0][n][v] = bv; acc[1][n][v] = bv; }
  }
#pragma unroll
  for (int c = 0; c < 12; ++c) {
    const int seg = c >> 2, k0 = (c & 3) * 32;
    const v16bf a0 = load_afrag(rp[0][seg] + k0, hoff);
    const v16bf a1 = load_afrag(rp[1][seg] + k0, hoff);
#pragma unroll
    for (int n = 0; n < 8; ++n) {
      const v16bf b = W1[(c * 8 + n) * 32 + lane];
      acc[0][n] = wmma_bf16(a0, b, acc[0][n]);
      acc[1][n] = wmma_bf16(a1, b, acc[1][n]);
    }
  }
#pragma unroll
  for (int s = 0; s < 2; ++s)
#pragma unroll
    for (int n = 0; n < 8; ++n)
#pragma unroll
      for (int v = 0; v < 8; ++v)
        hl[(s * 16 + v + half * 8) * 132 + n * 16 + lcol] = fmaxf(acc[s][n][v], 0.f);

  // ---- layer 2: 128 -> 128, B reused for 2 tiles ----
  v8f acc2[2][8];
#pragma unroll
  for (int n = 0; n < 8; ++n) {
    const float bv = bias2[n * 16 + lcol];
#pragma unroll
    for (int v = 0; v < 8; ++v) { acc2[0][n][v] = bv; acc2[1][n][v] = bv; }
  }
#pragma unroll
  for (int c = 0; c < 4; ++c) {
    const v16bf a0 = load_afrag(hl + lcol * 132 + c * 32, hoff);
    const v16bf a1 = load_afrag(hl + (16 + lcol) * 132 + c * 32, hoff);
#pragma unroll
    for (int n = 0; n < 8; ++n) {
      const v16bf b = W2[(c * 8 + n) * 32 + lane];
      acc2[0][n] = wmma_bf16(a0, b, acc2[0][n]);
      acc2[1][n] = wmma_bf16(a1, b, acc2[1][n]);
    }
  }

  // ---- layernorm + residual + store + fused segment-sum atomics ----
  float lsc[8], lbi[8];
#pragma unroll
  for (int n = 0; n < 8; ++n) { lsc[n] = lnsc[n * 16 + lcol]; lbi[n] = lnbi[n * 16 + lcol]; }

#pragma unroll
  for (int s = 0; s < 2; ++s) {
    float mu[8], rs[8];
#pragma unroll
    for (int v = 0; v < 8; ++v) {
      float sm = 0.f, s2 = 0.f;
#pragma unroll
      for (int n = 0; n < 8; ++n) { const float y = acc2[s][n][v]; sm += y; s2 += y * y; }
#pragma unroll
      for (int m = 1; m <= 8; m <<= 1) { sm += __shfl_xor(sm, m, 32); s2 += __shfl_xor(s2, m, 32); }
      const float m1 = sm * (1.f / 128.f);
      mu[v] = m1;
      rs[v] = rsqrtf(fmaxf(s2 * (1.f / 128.f) - m1 * m1, 0.f) + 1e-5f);
    }
#pragma unroll
    for (int v = 0; v < 8; ++v) {
      const int grow = r0 + s * 16 + v + half * 8;
      const bool ok = grow < rows;
      const int gr = ok ? grow : rows - 1;
      const int ga = (agg && ok) ? aggidx[gr] : 0;
#pragma unroll
      for (int n = 0; n < 8; ++n) {
        const int col = n * 16 + lcol;
        const float y = (acc2[s][n][v] - mu[v]) * rs[v] * lsc[n] + lbi[n];
        const float o = resid[(size_t)gr * 128 + col] + y;
        if (ok) {
          out[(size_t)gr * 128 + col] = o;
          if (agg) atomicAdd(&agg[(size_t)ga * 128 + col], o);
        }
      }
    }
  }
}

// ---------------------------------------------------------------------------
// Decoder: 128 -> 128 (relu, WMMA) -> 2 (scalar). One wave = 16 rows.
// ---------------------------------------------------------------------------
__global__ void __launch_bounds__(128)
mgn_dec(const float* __restrict__ x, const v16bf* __restrict__ W0,
        const float* __restrict__ b0, const float* __restrict__ W1,
        const float* __restrict__ b1, float* __restrict__ out, int rows) {
  __shared__ float lds[4][16 * 132];
  const int wave = threadIdx.x >> 5, lane = threadIdx.x & 31;
  const int tile = blockIdx.x * 4 + wave;
  const int r0 = tile * 16;
  if (r0 >= rows) return;
  const int half = lane >> 4, lcol = lane & 15, hoff = half * 8;
  int arow = r0 + lcol; if (arow >= rows) arow = rows - 1;
  float* hl = &lds[wave][0];

  v8f acc[8];
#pragma unroll
  for (int n = 0; n < 8; ++n) {
    const float bv = b0[n * 16 + lcol];
#pragma unroll
    for (int v = 0; v < 8; ++v) acc[n][v] = bv;
  }
  const float* rpx = x + (size_t)arow * 128;
#pragma unroll
  for (int c = 0; c < 4; ++c) {
    const v16bf a = load_afrag(rpx + c * 32, hoff);
#pragma unroll
    for (int n = 0; n < 8; ++n)
      acc[n] = wmma_bf16(a, W0[(c * 8 + n) * 32 + lane], acc[n]);
  }
#pragma unroll
  for (int n = 0; n < 8; ++n)
#pragma unroll
    for (int v = 0; v < 8; ++v)
      hl[(v + half * 8) * 132 + n * 16 + lcol] = fmaxf(acc[n][v], 0.f);
  // layer 1: 128 -> 2, lane = (row, col) with row=lane&15, col=lane>>4
  const int row = lcol, col = half;
  float s = b1[col];
#pragma unroll 8
  for (int k = 0; k < 128; ++k) s += hl[row * 132 + k] * W1[k * 2 + col];
  const int grow = r0 + row;
  if (grow < rows) out[(size_t)grow * 2 + col] = s;
}

// ---------------------------------------------------------------------------
// Host launcher
// ---------------------------------------------------------------------------
extern "C" void kernel_launch(void* const* d_in, const int* in_sizes, int n_in,
                              void* d_out, int out_size, void* d_ws, size_t ws_size,
                              hipStream_t stream) {
  (void)out_size; (void)ws_size;
  const float* pos   = (const float*)d_in[0];
  const float* mpos  = (const float*)d_in[1];
  const float* vel   = (const float*)d_in[2];
  const float* imass = (const float*)d_in[3];
  const float* nattr = (const float*)d_in[4];
  const int N  = in_sizes[0] / 2;
  const int* eidx = (const int*)d_in[n_in - 2];
  const int* widx = (const int*)d_in[n_in - 1];
  const int E  = in_sizes[n_in - 2] / 2;
  const int EW = in_sizes[n_in - 1] / 2;
  const int* src = eidx;       const int* dst = eidx + E;
  const int* sw  = widx;       const int* dw  = widx + EW;

  // params leaves in JAX pytree order (dict keys sorted), base index 5:
  // blocks[i] at 5+18i: edge{b0,b1,ln_bias,ln_scale,w0,w1}=+0..5,
  //                     node=+6..11, world=+12..17
  const int PB = 5;
  const int DEC = PB + 108;       // b0, b1, w0, w1
  const int ENC_MESH = DEC + 4;   // b0,b1,b2,ln_bias,ln_scale,w0,w1,w2
  const int ENC_NODE = ENC_MESH + 8;
  const int ENC_WORLD = ENC_NODE + 8;
  const int MASS_MEAN = ENC_WORLD + 8, MASS_STD = MASS_MEAN + 1;
  const int MESH_MEAN = MASS_STD + 1, MESH_STD = MESH_MEAN + 1;
  const int NODE_MEAN = MESH_STD + 1, NODE_STD = NODE_MEAN + 1;
  const int WORLD_MEAN = NODE_STD + 1, WORLD_STD = WORLD_MEAN + 1;
  auto P = [&](int i) { return (const float*)d_in[i]; };

  // workspace carve-out
  char* wsb = (char*)d_ws;
  size_t off = 0;
  auto alloc = [&](size_t bytes) -> void* {
    void* p = wsb + off;
    off = (off + bytes + 255) & ~(size_t)255;
    return p;
  };
  float* x    = (float*)alloc((size_t)N  * 128 * 4);
  float* me   = (float*)alloc((size_t)E  * 128 * 4);
  float* we   = (float*)alloc((size_t)EW * 128 * 4);
  float* aggm = (float*)alloc((size_t)N  * 128 * 4);
  float* aggw = (float*)alloc((size_t)N  * 128 * 4);
  float* n32  = (float*)alloc((size_t)N  * 32 * 4);
  float* m32  = (float*)alloc((size_t)E  * 32 * 4);
  float* w32  = (float*)alloc((size_t)EW * 32 * 4);
  auto allocPk = [&](int Kpad) { return (__bf16*)alloc((size_t)(Kpad / 32) * 8 * 32 * 16 * 2); };
  __bf16* pkB[6][3][2];
  for (int i = 0; i < 6; ++i)
    for (int j = 0; j < 3; ++j) { pkB[i][j][0] = allocPk(384); pkB[i][j][1] = allocPk(128); }
  __bf16* pkEN[3] = {allocPk(32), allocPk(128), allocPk(128)};   // enc_node
  __bf16* pkEM[3] = {allocPk(32), allocPk(128), allocPk(128)};   // enc_mesh
  __bf16* pkEWd[3] = {allocPk(32), allocPk(128), allocPk(128)};  // enc_world
  __bf16* pkDec = allocPk(128);

  auto cdiv = [](int a, int b) { return (a + b - 1) / b; };
  auto packW = [&](const float* W, __bf16* o, int Kpad, int Kact) {
    const int nth = (Kpad / 32) * 8 * 32;
    mgn_pack_w<<<cdiv(nth, 256), 256, 0, stream>>>(W, o, Kpad, Kact);
  };

  // 1) pack all weights (bf16 B-fragment order)
  for (int i = 0; i < 6; ++i) {
    const int bb = PB + 18 * i;
    packW(P(bb + 4),  pkB[i][0][0], 384, 384);  // edge  w0
    packW(P(bb + 5),  pkB[i][0][1], 128, 128);  // edge  w1
    packW(P(bb + 10), pkB[i][1][0], 384, 384);  // node  w0
    packW(P(bb + 11), pkB[i][1][1], 128, 128);  // node  w1
    packW(P(bb + 16), pkB[i][2][0], 384, 384);  // world w0
    packW(P(bb + 17), pkB[i][2][1], 128, 128);  // world w1
  }
  packW(P(ENC_NODE + 5),  pkEN[0], 32, 7);
  packW(P(ENC_NODE + 6),  pkEN[1], 128, 128);
  packW(P(ENC_NODE + 7),  pkEN[2], 128, 128);
  packW(P(ENC_MESH + 5),  pkEM[0], 32, 8);
  packW(P(ENC_MESH + 6),  pkEM[1], 128, 128);
  packW(P(ENC_MESH + 7),  pkEM[2], 128, 128);
  packW(P(ENC_WORLD + 5), pkEWd[0], 32, 4);
  packW(P(ENC_WORLD + 6), pkEWd[1], 128, 128);
  packW(P(ENC_WORLD + 7), pkEWd[2], 128, 128);
  packW(P(DEC + 2), pkDec, 128, 128);

  // 2) feature prep
  mgn_feat_node<<<cdiv(N, 256), 256, 0, stream>>>(vel, imass, nattr,
      P(NODE_MEAN), P(NODE_STD), P(MASS_MEAN), P(MASS_STD), n32, N);
  mgn_feat_mesh<<<cdiv(E, 256), 256, 0, stream>>>(pos, mpos, vel, src, dst,
      P(MESH_MEAN), P(MESH_STD), m32, E);
  mgn_feat_world<<<cdiv(EW, 256), 256, 0, stream>>>(pos, vel, sw, dw,
      P(WORLD_MEAN), P(WORLD_STD), w32, EW);

  // 3) encoders (WMMA, 16 rows/wave, 4 waves/block)
  const int gN  = cdiv(cdiv(N, 16), 4);
  const int gE  = cdiv(cdiv(E, 16), 4);
  const int gEw = cdiv(cdiv(EW, 16), 4);
  mgn_enc_mlp<<<gN, 128, 0, stream>>>(n32, (v16bf*)pkEN[0], (v16bf*)pkEN[1], (v16bf*)pkEN[2],
      P(ENC_NODE + 0), P(ENC_NODE + 1), P(ENC_NODE + 2),
      P(ENC_NODE + 4), P(ENC_NODE + 3), x, N);
  mgn_enc_mlp<<<gE, 128, 0, stream>>>(m32, (v16bf*)pkEM[0], (v16bf*)pkEM[1], (v16bf*)pkEM[2],
      P(ENC_MESH + 0), P(ENC_MESH + 1), P(ENC_MESH + 2),
      P(ENC_MESH + 4), P(ENC_MESH + 3), me, E);
  mgn_enc_mlp<<<gEw, 128, 0, stream>>>(w32, (v16bf*)pkEWd[0], (v16bf*)pkEWd[1], (v16bf*)pkEWd[2],
      P(ENC_WORLD + 0), P(ENC_WORLD + 1), P(ENC_WORLD + 2),
      P(ENC_WORLD + 4), P(ENC_WORLD + 3), we, EW);

  // 4) message-passing blocks (32 rows/wave, 2 waves/block)
  const int bN  = cdiv(cdiv(N, 32), 2);
  const int bE  = cdiv(cdiv(E, 32), 2);
  const int bEw = cdiv(cdiv(EW, 32), 2);
  for (int i = 0; i < 6; ++i) {
    const int bb = PB + 18 * i;
    mgn_zero2<<<cdiv(N * 128, 256), 256, 0, stream>>>(aggm, aggw, N * 128);
    // mesh-edge update (+ fused segment_sum into aggm[dst])
    mgn_block_mlp<<<bE, 64, 0, stream>>>(x, src, x, dst, me, me, me, aggm, dst,
        (v16bf*)pkB[i][0][0], (v16bf*)pkB[i][0][1],
        P(bb + 0), P(bb + 1), P(bb + 3), P(bb + 2), E);
    // world-edge update (+ fused segment_sum into aggw[dw])
    mgn_block_mlp<<<bEw, 64, 0, stream>>>(x, sw, x, dw, we, we, we, aggw, dw,
        (v16bf*)pkB[i][2][0], (v16bf*)pkB[i][2][1],
        P(bb + 12), P(bb + 13), P(bb + 15), P(bb + 14), EW);
    // node update
    mgn_block_mlp<<<bN, 64, 0, stream>>>(x, nullptr, aggm, nullptr, aggw, x, x,
        nullptr, nullptr,
        (v16bf*)pkB[i][1][0], (v16bf*)pkB[i][1][1],
        P(bb + 6), P(bb + 7), P(bb + 9), P(bb + 8), N);
  }

  // 5) decoder -> d_out (N x 2 f32)
  mgn_dec<<<gN, 128, 0, stream>>>(x, (v16bf*)pkDec, P(DEC + 0), P(DEC + 3),
                                  P(DEC + 1), (float*)d_out, N);
}
